// GGUFWeight_55293408968736
// MI455X (gfx1250) — compile-verified, matrix operand order checked by reading, not run on hardware
//
#include <hip/hip_runtime.h>
#include <stdint.h>

// ---------------------------------------------------------------------------
// q4_0 dequant GEMM for MI455X (gfx1250, wave32, WMMA).
//   out[32,4096] = x[32,14336] @ W^T + bias,  W stored as GGUF q4_0 with
//   every byte widened to int32 by the harness (132 MB weight stream).
// Memory-bound: ~140 MB @ 23.3 TB/s ~ 6 us. Strategy:
//   - stream qweight once with dword-aligned global_load_b128
//   - register dequant: v_perm byte pack + f16 exponent trick + v_pk_fma_f16
//   - v_wmma_f32_16x16x32_f16, one B tile shared by both batch-half WMMAs
//   - K split across the 8 waves of a block (2048 waves total for latency
//     hiding), partials reduced through LDS -> no global atomics, bias
//     folded into the single store, single kernel launch.
// ---------------------------------------------------------------------------

#define R_DIM   4096
#define IN_F    14336
#define BATCH   32
#define NB      448            // q4_0 blocks per row (IN_F/32)
#define ROW_I32 (NB * 18)      // 8064 int32 per weight row
#define SPLITK  8              // waves per block, each takes NB/SPLITK blocks
#define KB_PER  (NB / SPLITK)  // 56

typedef _Float16 v16h  __attribute__((ext_vector_type(16)));
typedef float    v8f   __attribute__((ext_vector_type(8)));
typedef _Float16 h2    __attribute__((ext_vector_type(2)));
typedef uint32_t u32x4 __attribute__((ext_vector_type(4), aligned(8)));

union FragH { v16h v; h2 h[8]; };

__device__ __forceinline__ h2 u32_as_h2(uint32_t u) {
  union { uint32_t u; h2 h; } c; c.u = u; return c.h;
}

__global__ __launch_bounds__(256) void q40_gemm(
    const float* __restrict__ x,       // [32, 14336] f32
    const int*   __restrict__ qw,      // [4096, 8064] int32 (one byte each)
    const float* __restrict__ scales,  // [4096, 448] f32
    const float* __restrict__ bias,    // [4096] f32
    float*       __restrict__ out) {   // [32, 4096] f32
  const int lane = threadIdx.x & 31;
  const int wv   = threadIdx.x >> 5;   // K-split id, 0..7
  const int g    = lane >> 4;          // half-wave group
  const int ln   = lane & 15;

  const int rbase = blockIdx.x * 16;   // 16 W rows per block (N tile)
  const int r     = rbase + ln;
  const int kb0   = wv * KB_PER;

  const int*   wrow = qw + (size_t)r * ROW_I32 + (size_t)kb0 * 18;
  const float* srow = scales + (size_t)r * NB + kb0;
  // A fragment layout (16-bit A 16x32): lanes 0-15 hold M=lane,
  //   halves 0..7 = K 8g..8g+7, halves 8..15 = K 16+8g..23+8g.
  const float* x0 = x + (size_t)ln * IN_F + (size_t)kb0 * 32 + g * 8;
  const float* x1 = x0 + (size_t)16 * IN_F;        // second batch half

  v8f acc0 = {}; v8f acc1 = {};
  const uint32_t C64 = 0x64646464u;   // f16 bits of 1024.0 per byte-pair
  const uint32_t M4  = 0x0F0F0F0Fu;
  const h2 kNeg1032 = {(_Float16)(-1032.0f), (_Float16)(-1032.0f)};

#pragma unroll 2
  for (int kb = 0; kb < KB_PER; ++kb) {
    // ---- stream weights: qs bytes 8g..8g+7 as int32s (dword aligned) ----
    u32x4 qA = *(const u32x4*)(wrow + 2 + 8 * g);  // bytes 8g+0..3
    u32x4 qB = *(const u32x4*)(wrow + 6 + 8 * g);  // bytes 8g+4..7
    __builtin_prefetch(wrow + 18 * 8, 0, 1);       // global_prefetch_b8

    // ---- A fragments: f32 -> f16 ----
    float4 a0 = *(const float4*)(x0);
    float4 a1 = *(const float4*)(x0 + 4);
    float4 a2 = *(const float4*)(x0 + 16);
    float4 a3 = *(const float4*)(x0 + 20);
    float4 b0 = *(const float4*)(x1);
    float4 b1 = *(const float4*)(x1 + 4);
    float4 b2 = *(const float4*)(x1 + 16);
    float4 b3 = *(const float4*)(x1 + 20);
    FragH A0, A1;
    A0.h[0] = (h2){(_Float16)a0.x, (_Float16)a0.y};
    A0.h[1] = (h2){(_Float16)a0.z, (_Float16)a0.w};
    A0.h[2] = (h2){(_Float16)a1.x, (_Float16)a1.y};
    A0.h[3] = (h2){(_Float16)a1.z, (_Float16)a1.w};
    A0.h[4] = (h2){(_Float16)a2.x, (_Float16)a2.y};
    A0.h[5] = (h2){(_Float16)a2.z, (_Float16)a2.w};
    A0.h[6] = (h2){(_Float16)a3.x, (_Float16)a3.y};
    A0.h[7] = (h2){(_Float16)a3.z, (_Float16)a3.w};
    A1.h[0] = (h2){(_Float16)b0.x, (_Float16)b0.y};
    A1.h[1] = (h2){(_Float16)b0.z, (_Float16)b0.w};
    A1.h[2] = (h2){(_Float16)b1.x, (_Float16)b1.y};
    A1.h[3] = (h2){(_Float16)b1.z, (_Float16)b1.w};
    A1.h[4] = (h2){(_Float16)b2.x, (_Float16)b2.y};
    A1.h[5] = (h2){(_Float16)b2.z, (_Float16)b2.w};
    A1.h[6] = (h2){(_Float16)b3.x, (_Float16)b3.y};
    A1.h[7] = (h2){(_Float16)b3.z, (_Float16)b3.w};

    // ---- dequant: pack byte-valued int32s, split nibbles, f16 trick ----
    uint32_t bp0 = __builtin_amdgcn_perm(qA.y, qA.x, 0x0C0C0400u) |
                   __builtin_amdgcn_perm(qA.w, qA.z, 0x04000C0Cu);
    uint32_t bp1 = __builtin_amdgcn_perm(qB.y, qB.x, 0x0C0C0400u) |
                   __builtin_amdgcn_perm(qB.w, qB.z, 0x04000C0Cu);
    uint32_t nl0 =  bp0       & M4;   // low nibbles  -> K 8g..8g+3
    uint32_t nl1 =  bp1       & M4;   //              -> K 8g+4..8g+7
    uint32_t nh0 = (bp0 >> 4) & M4;   // high nibbles -> K 16+8g..
    uint32_t nh1 = (bp1 >> 4) & M4;

    _Float16 dh = (_Float16)srow[kb];
    h2 dd; dd.x = dh; dd.y = dh;
    h2 bb = dd * kNeg1032;            // -1032*d, so (1024+n)*d+bb = (n-8)*d

    // B fragment (16-bit B 32x16, mirroring A layout): lanes 0-15 N=lane,
    // halves 0..7 = K 8g..8g+7, halves 8..15 = K 16+8g..23+8g.
    FragH B;
    B.h[0] = u32_as_h2(__builtin_amdgcn_perm(C64, nl0, 0x04010400u)) * dd + bb;
    B.h[1] = u32_as_h2(__builtin_amdgcn_perm(C64, nl0, 0x04030402u)) * dd + bb;
    B.h[2] = u32_as_h2(__builtin_amdgcn_perm(C64, nl1, 0x04010400u)) * dd + bb;
    B.h[3] = u32_as_h2(__builtin_amdgcn_perm(C64, nl1, 0x04030402u)) * dd + bb;
    B.h[4] = u32_as_h2(__builtin_amdgcn_perm(C64, nh0, 0x04010400u)) * dd + bb;
    B.h[5] = u32_as_h2(__builtin_amdgcn_perm(C64, nh0, 0x04030402u)) * dd + bb;
    B.h[6] = u32_as_h2(__builtin_amdgcn_perm(C64, nh1, 0x04010400u)) * dd + bb;
    B.h[7] = u32_as_h2(__builtin_amdgcn_perm(C64, nh1, 0x04030402u)) * dd + bb;

    // ---- two WMMAs share the dequantized B tile ----
    acc0 = __builtin_amdgcn_wmma_f32_16x16x32_f16(
        false, A0.v, false, B.v, (short)0, acc0, false, false);
    acc1 = __builtin_amdgcn_wmma_f32_16x16x32_f16(
        false, A1.v, false, B.v, (short)0, acc1, false, false);

    wrow += 18; x0 += 32; x1 += 32;
  }

  // ---- LDS reduction of the 8 K-split partials (no global atomics) ----
  // C/D layout: acc VGPR i -> M = i + 8*g (acc1: +16), N = ln.
  __shared__ float red[SPLITK][2][32][8];   // 16 KB
#pragma unroll
  for (int i = 0; i < 8; ++i) {
    red[wv][0][lane][i] = acc0[i];
    red[wv][1][lane][i] = acc1[i];
  }
  __syncthreads();

  // 512 outputs (32 batch rows x 16 cols), 2 per thread.
#pragma unroll
  for (int e = threadIdx.x; e < BATCH * 16; e += 256) {
    const int mm   = e >> 4;          // batch row 0..31
    const int n    = e & 15;          // column within tile
    const int half = mm >> 4;         // acc0 / acc1
    const int m16  = mm & 15;
    const int rl   = ((m16 & 8) ? 16 : 0) + n;   // source lane
    const int i    = m16 & 7;                    // source VGPR index
    float s = 0.0f;
#pragma unroll
    for (int w = 0; w < SPLITK; ++w) s += red[w][half][rl][i];
    out[(size_t)mm * R_DIM + rbase + n] = s + bias[rbase + n];
  }
}

extern "C" void kernel_launch(void* const* d_in, const int* in_sizes, int n_in,
                              void* d_out, int out_size, void* d_ws, size_t ws_size,
                              hipStream_t stream) {
  const float* x      = (const float*)d_in[0];
  const int*   qw     = (const int*)d_in[1];
  const float* scales = (const float*)d_in[2];
  const float* bias   = (const float*)d_in[3];
  float* out = (float*)d_out;

  // 256 blocks (one 16-column tile each) x 256 threads (8 K-split waves).
  q40_gemm<<<R_DIM / 16, 256, 0, stream>>>(x, qw, scales, bias, out);
}
